// CouplingLayer_41635412968146
// MI455X (gfx1250) — matrix operand order repaired and checked by reading, not compiled, not run on hardware
//
#include <hip/hip_runtime.h>
#include <hip/hip_bf16.h>
#include <math.h>

typedef __attribute__((ext_vector_type(16))) _Float16 v16h;
typedef __attribute__((ext_vector_type(8)))  float    v8f;

#define WAVES_PER_BLOCK 8
#define PAIRS_PER_WAVE  2
#define ROWS_PER_PAIR   32
#define ROWS_PER_WAVE   (PAIRS_PER_WAVE * ROWS_PER_PAIR)          // 64
#define ROWS_PER_BLOCK  (WAVES_PER_BLOCK * ROWS_PER_WAVE)         // 512
#define HDIM 64     // hidden width (WMMA K = 64 -> 2 chained K=32 WMMAs)
#define TDIM 96     // t_feat width
#define NMAX 4      // batch (n) count
#define OUTD 10     // map-net output width (padded to 16 for WMMA N)

// Fast softplus on the hardware trans pipe: v_exp_f32 + v_log_f32.
__device__ __forceinline__ float softplus_f(float x) {
    return fmaxf(x, 0.0f) + __logf(1.0f + __expf(-fabsf(x)));
}

// Fast tanh: hardware v_tanh_f32 when the builtin exists, else 1 trans op + rcp.
__device__ __forceinline__ float tanh_f(float x) {
#if __has_builtin(__builtin_amdgcn_tanhf)
    return __builtin_amdgcn_tanhf(x);
#else
    const float e = __expf(2.0f * fabsf(x));
    return copysignf(1.0f - 2.0f / (e + 1.0f), x);
#endif
}

// One 16-row WMMA tile: folded layer-1 (2 FMAs/hidden unit) -> f16 A-fragments
// -> D(16x16) = A(16x64) x W2(64x16) on the matrix pipe.
__device__ __forceinline__ v8f compute_tile(
    unsigned tileBase, int lane, unsigned R, unsigned rowsPerN,
    const float* __restrict__ inp,
    const float* sW1, const float* sBase,
    v16h bfrag0, v16h bfrag1,
    float& x0o, float& x1o, float& zqo)
{
    const int m = lane & 15;
    unsigned row = tileBase + (unsigned)m;
    const unsigned rowC = (row < R) ? row : (R - 1u);

    const float x0 = inp[rowC * 3 + 0];
    const float x1 = inp[rowC * 3 + 1];
    const float zq = inp[rowC * 3 + 2];
    x0o = x0; x1o = x1; zqo = zq;

    const float c0 = tanh_f(x0);
    const float c1 = tanh_f(x1);
    const unsigned n = rowC / rowsPerN;
    const float* bs = &sBase[n * HDIM];

    // A 16x32 f16 layout: half j of lane -> K = j + (lane>>4)*16
    const int koff = (lane >> 4) * 16;
    v16h a0, a1;
    #pragma unroll
    for (int j = 0; j < 16; ++j) {
        const int kl = j + koff;   // K in [0,32)
        const int kh = kl + 32;                            // K in [32,64)
        float h0 = fmaf(c0, sW1[kl], fmaf(c1, sW1[HDIM + kl], bs[kl]));
        float h1 = fmaf(c0, sW1[kh], fmaf(c1, sW1[HDIM + kh], bs[kh]));
        a0[j] = (_Float16)fmaxf(h0, 0.0f);
        a1[j] = (_Float16)fmaxf(h1, 0.0f);
    }

    v8f acc = {0.f, 0.f, 0.f, 0.f, 0.f, 0.f, 0.f, 0.f};
    acc = __builtin_amdgcn_wmma_f32_16x16x32_f16(
              false, a0, false, bfrag0, (short)0, acc, false, false);
    acc = __builtin_amdgcn_wmma_f32_16x16x32_f16(
              false, a1, false, bfrag1, (short)0, acc, false, false);
    return acc;
}

__global__ __launch_bounds__(256) void coupling_layer_kernel(
    const float* __restrict__ inp,    // (R, 3)
    const float* __restrict__ tfeat,  // (N, 96)
    const float* __restrict__ W1,     // (98, 64) row-major
    const float* __restrict__ b1,     // (64)
    const float* __restrict__ W2,     // (64, 10) row-major
    const float* __restrict__ b2,     // (10)
    float* __restrict__ out,          // (R, 3)
    unsigned R, unsigned rowsPerN)
{
    __shared__ float sW1[2 * HDIM];                   // W1 cond-rows 0,1
    __shared__ float sBase[NMAX * HDIM];              // folded t_feat@W1[2:] + b1 per n
    __shared__ float sD[WAVES_PER_BLOCK][ROWS_PER_PAIR * 16];  // 32x16 D transpose buffer

    const int tid  = threadIdx.x;
    const int lane = tid & 31;
    const int wave = tid >> 5;

    // ---- Block prolog: stage W1 cond-rows and fold the t-feature GEMM (tiny) ----
    if (tid < 2 * HDIM) sW1[tid] = W1[tid];
    {
        const int n = tid >> 6;              // 256 threads = 4 x 64 exactly
        const int h = tid & (HDIM - 1);
        float acc = b1[h];
        #pragma unroll 8
        for (int t = 0; t < TDIM; ++t)
            acc = fmaf(tfeat[n * TDIM + t], W1[(2 + t) * HDIM + h], acc);
        sBase[tid] = acc;
    }
    __syncthreads();

    // ---- B fragments for W2 (64x10 -> 64x16 zero-padded), constant per wave ----
    // B 32x16 f16 layout: lanes 0-15 hold K=0..15 (N=lane), lanes 16-31 hold K=16..31.
    const int col   = lane & 15;
    const int khalf = (lane >> 4) * 16;
    v16h bfrag0, bfrag1;
    #pragma unroll
    for (int j = 0; j < 16; ++j) {
        const int k0 = khalf + j;
        const int k1 = 32 + khalf + j;
        bfrag0[j] = (col < OUTD) ? (_Float16)W2[k0 * OUTD + col] : (_Float16)0.0f;
        bfrag1[j] = (col < OUTD) ? (_Float16)W2[k1 * OUTD + col] : (_Float16)0.0f;
    }

    // Hoist b2 (uniform) into registers once.
    float bb[OUTD];
    #pragma unroll
    for (int k = 0; k < OUTD; ++k) bb[k] = b2[k];

    const unsigned blockBase = (unsigned)blockIdx.x * ROWS_PER_BLOCK;
    float* dbuf = sD[wave];
    const int rbase = (lane >> 4) * 8;

    for (int p = 0; p < PAIRS_PER_WAVE; ++p) {
        const unsigned pairBase = blockBase + (unsigned)wave * ROWS_PER_WAVE
                                            + (unsigned)p * ROWS_PER_PAIR;

        // ---- Tile A: rows [pairBase, pairBase+16) ----
        float xa0, xa1, za;
        v8f accA = compute_tile(pairBase, lane, R, rowsPerN, inp, sW1, sBase,
                                bfrag0, bfrag1, xa0, xa1, za);
        #pragma unroll
        for (int r = 0; r < 8; ++r)
            dbuf[(rbase + r) * 16 + col] = accA[r];

        // ---- Tile B: rows [pairBase+16, pairBase+32) ----
        float xb0, xb1, zb;
        v8f accB = compute_tile(pairBase + 16, lane, R, rowsPerN, inp, sW1, sBase,
                                bfrag0, bfrag1, xb0, xb1, zb);
        #pragma unroll
        for (int r = 0; r < 8; ++r)
            dbuf[(16 + rbase + r) * 16 + col] = accB[r];

        // Lane L will own row pairBase+L: lanes<16 keep tile-A inputs, >=16 tile-B.
        const float x0 = (lane < 16) ? xa0 : xb0;
        const float x1 = (lane < 16) ? xa1 : xb1;
        const float zq = (lane < 16) ? za  : zb;

        __syncthreads();

        // ---- Epilogue: all 32 lanes active, one row each ----
        const unsigned row = pairBase + (unsigned)lane;
        if (row < R) {
            float d[OUTD];
            #pragma unroll
            for (int k = 0; k < OUTD; ++k)
                d[k] = softplus_f(dbuf[lane * 16 + k] + bb[k]) + 1e-4f;

            const float dxl2 = d[0], dxl1 = d[1], dxr1 = d[2], dxr2 = d[3];
            const float dyl2 = d[4], dyl1 = d[5], dyr1 = d[6], dyr2 = d[7];
            const float kl = 2.0f * d[8], kr = 2.0f * d[9];

            const float xL1 = -dxl1,        xL2 = -dxl1 - dxl2;
            const float yL1 = -dyl1,        yL2 = -dyl1 - dyl2;
            const float xR1 =  dxr1,        xR2 =  dxr1 + dxr2;
            const float yR1 =  dyr1,        yR2 =  dyr1 + dyr2;
            const float xR3 = xR2 + 10000.0f, xL3 = xL2 - 10000.0f;
            const float yR3 = yR2 + kr * 10000.0f, yL3 = yL2 - kl * 10000.0f;

            const float ax[6] = {xL3, xL2, xL1, xR1, xR2, xR3};
            const float ay[6] = {yL3, yL2, yL1, yR1, yR2, yR3};

            const float qx = fminf(fmaxf(zq, ax[0] * 0.99f), ax[5] * 0.99f);
            float zi = 0.0f;
            #pragma unroll
            for (int i = 0; i < 5; ++i) {
                const bool inr = (qx >= ax[i]) && (qx < ax[i + 1]);
                const float slope = __fdividef(ay[i + 1] - ay[i], ax[i + 1] - ax[i]);
                const float v = fmaf(slope, qx - ax[i], ay[i]);
                zi += inr ? v : 0.0f;
            }

            out[row * 3 + 0] = x0;
            out[row * 3 + 1] = x1;
            out[row * 3 + 2] = zi;
        }
        __syncthreads();
    }
}

extern "C" void kernel_launch(void* const* d_in, const int* in_sizes, int n_in,
                              void* d_out, int out_size, void* d_ws, size_t ws_size,
                              hipStream_t stream) {
    const float* inp   = (const float*)d_in[0];
    const float* tfeat = (const float*)d_in[1];
    const float* W1    = (const float*)d_in[2];
    const float* b1    = (const float*)d_in[3];
    const float* W2    = (const float*)d_in[4];
    const float* b2    = (const float*)d_in[5];
    float* out = (float*)d_out;

    const unsigned R = (unsigned)(in_sizes[0] / 3);        // N*P*S rows
    const unsigned N = (unsigned)(in_sizes[1] / TDIM);     // batch count
    const unsigned rowsPerN = R / (N ? N : 1);

    const unsigned blocks = (R + ROWS_PER_BLOCK - 1) / ROWS_PER_BLOCK;
    coupling_layer_kernel<<<blocks, 256, 0, stream>>>(
        inp, tfeat, W1, b1, W2, b2, out, R, rowsPerN);
}
